// WaveGlow_62895501083230
// MI455X (gfx1250) — compile-verified, matrix operand order checked by reading, not run on hardware
//
#include <hip/hip_runtime.h>
#include <hip/hip_bf16.h>
#include <math.h>

// ---------------------------------------------------------------------------
// WaveGlow inference on gfx1250 (MI455X).
// Heavy GEMMs (cond 640->512, 3x dilated-conv taps 256->512, res/skip 256->512)
// run on v_wmma_f32_16x16x32_bf16 with fp32 accumulation, software-pipelined
// (double-buffered fragments, loads one K-chunk ahead of the matrix pipe).
// Weights are transposed+converted fp32->bf16 per flow via an LDS-tiled
// kernel; activations feeding WMMA are kept in bf16 row-major.
// ---------------------------------------------------------------------------

typedef __attribute__((ext_vector_type(16))) __bf16 v16bf;
typedef __attribute__((ext_vector_type(8)))  float  v8f;

#define TP    4096     // time steps after grouping
#define TFULL 32768    // upsampled length
#define NC    256
#define CMEL  640      // N_MEL * N_GROUP

// ---------------------------------------------------------------------------
// Upsample (conv_transpose, VALID, stride 256, K=1024) + grouping, write bf16
// s[tg, c*8+g] = s_full[tg*8+g, c], row-major (4096, 640) -> WMMA A layout.
// ---------------------------------------------------------------------------
__global__ __launch_bounds__(256) void upsample_kernel(
    const float* __restrict__ spect, const float* __restrict__ upw,
    const float* __restrict__ upb, __bf16* __restrict__ s_bf)
{
  int i = blockIdx.x * 256 + threadIdx.x;
  if (i >= TFULL * 80) return;
  int t = i / 80, co = i % 80;
  int mLo = (t >= 1023) ? ((t - 1023 + 255) >> 8) : 0;
  int mHi = t >> 8; if (mHi > 127) mHi = 127;
  float acc = upb[co];
  for (int m = mLo; m <= mHi; ++m) {
    int kk = t - (m << 8);                       // kernel tap index
    const float* wrow = upw + (size_t)kk * 6400; // (1024, 80, 80)
    const float* srow = spect + m * 80;
    #pragma unroll 8
    for (int ci = 0; ci < 80; ++ci) acc += srow[ci] * wrow[ci * 80 + co];
  }
  int tg = t >> 3, g = t & 7;
  s_bf[(size_t)tg * CMEL + co * 8 + g] = (__bf16)acc;
}

// ---------------------------------------------------------------------------
// LDS-tiled transpose + fp32->bf16 convert: src (K,N) row-major -> dst (N,K).
// ---------------------------------------------------------------------------
__global__ __launch_bounds__(256) void transpose_cvt_kernel(
    const float* __restrict__ src, __bf16* __restrict__ dst, int K, int N)
{
  __shared__ __bf16 tile[32][33];
  int kb = blockIdx.x * 32, nb = blockIdx.y * 32;
  int tx = threadIdx.x & 31, ty = threadIdx.x >> 5;   // 32 x 8
  #pragma unroll
  for (int i = 0; i < 32; i += 8)
    tile[ty + i][tx] = (__bf16)src[(size_t)(kb + ty + i) * N + (nb + tx)];
  __syncthreads();
  #pragma unroll
  for (int i = 0; i < 32; i += 8)
    dst[(size_t)(nb + ty + i) * K + (kb + tx)] = tile[tx][ty + i];
}

// ---------------------------------------------------------------------------
// WMMA GEMM: C(T,N) (+)= A(T,K)bf16 @ Wt(N,K)bf16 [+ bias].
// Block = 8 waves: wave (wm,wn) -> tile 32M x 64N; block tile 128x128.
// K is a multiple of 64 -> k-loop unrolled 2x with ping-pong fragment sets so
// loads stay one 32-chunk ahead of the WMMAs. OOB conv-tap rows use a zeroed
// scratch page with stride 0 (no divergence in the hot loop).
// ---------------------------------------------------------------------------
__global__ __launch_bounds__(256) void gemm_bf16_wmma(
    const __bf16* __restrict__ A, const __bf16* __restrict__ Wt,
    const float* __restrict__ bias, float* __restrict__ C,
    const __bf16* __restrict__ zpage,
    int T, int K, int N, int rowShift, int accum)
{
  const int lane = threadIdx.x & 31;
  const int wave = threadIdx.x >> 5;
  const int wm = wave & 3, wn = wave >> 2;
  const int mBase = blockIdx.x * 128 + wm * 32;
  const int nBase = blockIdx.y * 128 + wn * 64;
  const int colL  = lane & 15;
  const int kHalf = (lane >> 4) * 16;       // lanes 16-31 hold K+16..K+31

  const int r0 = mBase + colL + rowShift;
  const int r1 = r0 + 16;
  const bool ok0 = (r0 >= 0) && (r0 < T);
  const bool ok1 = (r1 >= 0) && (r1 < T);
  const __bf16* p0 = ok0 ? (A + (size_t)r0 * K + kHalf) : zpage;
  const __bf16* p1 = ok1 ? (A + (size_t)r1 * K + kHalf) : zpage;
  const int s0 = ok0 ? 32 : 0;              // per-chunk pointer stride (elems)
  const int s1 = ok1 ? 32 : 0;
  const __bf16* pb = Wt + (size_t)(nBase + colL) * K + kHalf;
  const size_t bS = (size_t)16 * K;

  v8f acc[2][4];
  #pragma unroll
  for (int mi = 0; mi < 2; ++mi)
    #pragma unroll
    for (int j = 0; j < 4; ++j) acc[mi][j] = (v8f){};

  // prologue: chunk 0 -> X set
  v16bf aX0 = *(const v16bf*)p0;
  v16bf aX1 = *(const v16bf*)p1;
  v16bf bX[4];
  #pragma unroll
  for (int j = 0; j < 4; ++j) bX[j] = *(const v16bf*)(pb + j * bS);
  p0 += s0; p1 += s1; pb += 32;

  for (int k = 0; k < K; k += 64) {
    // load chunk k+32 -> Y set (always exists: K % 64 == 0)
    v16bf aY0 = *(const v16bf*)p0;
    v16bf aY1 = *(const v16bf*)p1;
    v16bf bY[4];
    #pragma unroll
    for (int j = 0; j < 4; ++j) bY[j] = *(const v16bf*)(pb + j * bS);
    p0 += s0; p1 += s1; pb += 32;
    __builtin_prefetch((const void*)p0, 0, 1);   // global_prefetch_b8

    #pragma unroll
    for (int j = 0; j < 4; ++j) {
      acc[0][j] = __builtin_amdgcn_wmma_f32_16x16x32_bf16(false, aX0, false, bX[j], (short)0, acc[0][j], false, false);
      acc[1][j] = __builtin_amdgcn_wmma_f32_16x16x32_bf16(false, aX1, false, bX[j], (short)0, acc[1][j], false, false);
    }
    if (k + 64 < K) {   // load chunk k+64 -> X set while Y computes
      aX0 = *(const v16bf*)p0;
      aX1 = *(const v16bf*)p1;
      #pragma unroll
      for (int j = 0; j < 4; ++j) bX[j] = *(const v16bf*)(pb + j * bS);
      p0 += s0; p1 += s1; pb += 32;
    }
    #pragma unroll
    for (int j = 0; j < 4; ++j) {
      acc[0][j] = __builtin_amdgcn_wmma_f32_16x16x32_bf16(false, aY0, false, bY[j], (short)0, acc[0][j], false, false);
      acc[1][j] = __builtin_amdgcn_wmma_f32_16x16x32_bf16(false, aY1, false, bY[j], (short)0, acc[1][j], false, false);
    }
  }

  // C layout: lane -> col = base + lane%16 ; VGPR r -> row = r (+8 lanes>=16)
  const int rOff = (lane >> 4) * 8;
  #pragma unroll
  for (int mi = 0; mi < 2; ++mi) {
    #pragma unroll
    for (int s = 0; s < 4; ++s) {
      int col = nBase + s * 16 + colL;
      float bv = bias ? bias[col] : 0.0f;
      #pragma unroll
      for (int r = 0; r < 8; ++r) {
        size_t idx = (size_t)(mBase + mi * 16 + rOff + r) * N + col;
        float v = acc[mi][s][r] + bv;
        if (accum) v += C[idx];
        C[idx] = v;
      }
    }
  }
}

// ---------------------------------------------------------------------------
// Elementwise / small kernels
// ---------------------------------------------------------------------------
__global__ void zero_page_kernel(__bf16* __restrict__ z)
{
  z[threadIdx.x] = (__bf16)0.0f;   // 128 threads -> 256 B of zeros
}

__global__ __launch_bounds__(256) void gate_kernel(
    const float* __restrict__ z, __bf16* __restrict__ acts)
{
  int i = blockIdx.x * 256 + threadIdx.x;
  if (i >= TP * NC) return;
  int t = i >> 8, c = i & 255;
  float zt = z[(size_t)t * 512 + c];
  float zs = z[(size_t)t * 512 + 256 + c];
  acts[i] = (__bf16)(tanhf(zt) * (1.0f / (1.0f + __expf(-zs))));
}

__global__ __launch_bounds__(256) void resskip_kernel(
    const float* __restrict__ rs, float* __restrict__ x, __bf16* __restrict__ xb,
    float* __restrict__ out, int first, int last)
{
  int i = blockIdx.x * 256 + threadIdx.x;
  if (i >= TP * NC) return;
  int t = i >> 8, c = i & 255;
  if (last) { out[i] += rs[(size_t)t * 256 + c]; return; }
  float xn = x[i] + rs[(size_t)t * 512 + c];
  x[i] = xn; xb[i] = (__bf16)xn;
  float sk = rs[(size_t)t * 512 + 256 + c];
  out[i] = first ? sk : out[i] + sk;
}

__global__ __launch_bounds__(256) void start_kernel(
    const float* __restrict__ audio, const float* __restrict__ w,
    const float* __restrict__ b, int nh, float* __restrict__ x,
    __bf16* __restrict__ xb)
{
  int i = blockIdx.x * 256 + threadIdx.x;
  if (i >= TP * NC) return;
  int t = i >> 8, o = i & 255;
  float s = b[o];
  for (int c = 0; c < nh; ++c) s += audio[t * 8 + c] * w[c * 256 + o];
  x[i] = s; xb[i] = (__bf16)s;
}

__global__ __launch_bounds__(256) void end_kernel(
    const float* __restrict__ out, const float* __restrict__ w,
    const float* __restrict__ b, int n2, float* __restrict__ oend)
{
  int i = blockIdx.x * 256 + threadIdx.x;
  if (i >= TP * n2) return;
  int t = i / n2, j = i % n2;
  float s = b[j];
  const float* orow = out + (size_t)t * 256;
  #pragma unroll 8
  for (int c = 0; c < 256; ++c) s += orow[c] * w[c * n2 + j];
  oend[t * 16 + j] = s;
}

__global__ __launch_bounds__(256) void coupling_kernel(
    float* __restrict__ audio, const float* __restrict__ oend, int nh)
{
  int i = blockIdx.x * 256 + threadIdx.x;
  if (i >= TP * nh) return;
  int t = i / nh, j = i % nh;
  float a = audio[t * 8 + nh + j];
  audio[t * 8 + nh + j] = (a - oend[t * 16 + j]) * __expf(-oend[t * 16 + nh + j]);
}

__global__ void invert_kernel(const float* __restrict__ M, float* __restrict__ Minv, int n)
{
  if (threadIdx.x != 0 || blockIdx.x != 0) return;
  float a[8][16];
  for (int r = 0; r < n; ++r)
    for (int c = 0; c < n; ++c) { a[r][c] = M[r * n + c]; a[r][n + c] = (r == c) ? 1.0f : 0.0f; }
  for (int col = 0; col < n; ++col) {
    int p = col;
    for (int r = col + 1; r < n; ++r) if (fabsf(a[r][col]) > fabsf(a[p][col])) p = r;
    for (int c = 0; c < 2 * n; ++c) { float tm = a[col][c]; a[col][c] = a[p][c]; a[p][c] = tm; }
    float inv = 1.0f / a[col][col];
    for (int c = 0; c < 2 * n; ++c) a[col][c] *= inv;
    for (int r = 0; r < n; ++r) if (r != col) {
      float f = a[r][col];
      for (int c = 0; c < 2 * n; ++c) a[r][c] -= f * a[col][c];
    }
  }
  for (int r = 0; r < n; ++r)
    for (int c = 0; c < n; ++c) Minv[r * n + c] = a[r][n + c];
}

__global__ __launch_bounds__(256) void permute_kernel(
    float* __restrict__ audio, const float* __restrict__ Minv, int n,
    const float* __restrict__ zearly, int prepend)
{
  int t = blockIdx.x * 256 + threadIdx.x;
  if (t >= TP) return;
  float row[8], nv[8];
  for (int r = 0; r < n; ++r) row[r] = audio[t * 8 + r];
  for (int c = 0; c < n; ++c) {
    float s = 0.0f;
    for (int r = 0; r < n; ++r) s += row[r] * Minv[r * n + c];
    nv[c] = s;
  }
  if (prepend) {
    audio[t * 8 + 0] = zearly[t * 2 + 0];
    audio[t * 8 + 1] = zearly[t * 2 + 1];
    for (int c = 0; c < n; ++c) audio[t * 8 + 2 + c] = nv[c];
  } else {
    for (int c = 0; c < n; ++c) audio[t * 8 + c] = nv[c];
  }
}

__global__ __launch_bounds__(256) void init_audio_kernel(
    const float* __restrict__ z0, float* __restrict__ audio)
{
  int i = blockIdx.x * 256 + threadIdx.x;
  if (i >= TP * 8) return;
  int t = i >> 3, c = i & 7;
  audio[i] = (c < 4) ? z0[t * 4 + c] : 0.0f;
}

__global__ __launch_bounds__(256) void final_copy_kernel(
    const float* __restrict__ audio, float* __restrict__ out)
{
  int i = blockIdx.x * 256 + threadIdx.x;
  if (i < TP * 8) out[i] = audio[i];
}

// ---------------------------------------------------------------------------
// Host-side orchestration
// ---------------------------------------------------------------------------
extern "C" void kernel_launch(void* const* d_in, const int* in_sizes, int n_in,
                              void* d_out, int out_size, void* d_ws, size_t ws_size,
                              hipStream_t stream) {
  (void)in_sizes; (void)n_in; (void)out_size; (void)ws_size;

  // --- input pointer map (jax pytree flatten: dicts sorted by key) ---
  // 0:spect 1:z0 2:z_early8 3:z_early4 then 12 flows x 53, then up_b, up_w.
  auto F = [&](int idx) { return (const float*)d_in[idx]; };
  const float* spect = F(0);
  const float* z0    = F(1);
  const float* ze8   = F(2);
  const float* ze4   = F(3);
  auto fb = [&](int f) { return 4 + f * 53; };
  // per-flow: cond_b[i]=+i cond_w[i]=+8+i end_b=+16 end_w=+17 in_b[i]=+18+i
  //           in_w[i]=+26+i inv_w=+34 rs_b[i]=+35+i rs_w[i]=+43+i start_b=+51 start_w=+52
  const float* up_b = F(4 + 12 * 53);
  const float* up_w = F(4 + 12 * 53 + 1);

  // --- workspace layout ---
  char* ws = (char*)d_ws;
  size_t off = 0;
  auto alloc = [&](size_t bytes) { size_t o = off; off = (off + bytes + 255) & ~(size_t)255; return o; };
  __bf16* S_BF   = (__bf16*)(ws + alloc((size_t)TP * CMEL * 2));   // cond activations bf16
  float*  X_F    = (float*) (ws + alloc((size_t)TP * NC * 4));
  __bf16* X_BF   = (__bf16*)(ws + alloc((size_t)TP * NC * 2));
  float*  Z      = (float*) (ws + alloc((size_t)TP * 512 * 4));
  __bf16* ACT_BF = (__bf16*)(ws + alloc((size_t)TP * NC * 2));
  float*  RS     = (float*) (ws + alloc((size_t)TP * 512 * 4));
  float*  OUTF   = (float*) (ws + alloc((size_t)TP * NC * 4));
  __bf16* WSLAB  = (__bf16*)(ws + alloc((size_t)8 * 851968 * 2));  // per-flow bf16 weights
  float*  OEND   = (float*) (ws + alloc((size_t)TP * 16 * 4));
  float*  AUDIO  = (float*) (ws + alloc((size_t)TP * 8 * 4));
  float*  INVM   = (float*) (ws + alloc(256));
  __bf16* ZPAGE  = (__bf16*)(ws + alloc(256));                     // zeros for OOB rows

  auto blocks1d = [](size_t n) { return dim3((unsigned)((n + 255) / 256)); };

  zero_page_kernel<<<1, 128, 0, stream>>>(ZPAGE);
  // upsample + grouping -> S_BF
  upsample_kernel<<<blocks1d((size_t)TFULL * 80), 256, 0, stream>>>(spect, up_w, up_b, S_BF);
  // audio <- z0 (width 4, stride 8)
  init_audio_kernel<<<blocks1d((size_t)TP * 8), 256, 0, stream>>>(z0, AUDIO);

  auto gemm = [&](const __bf16* A, int K, const __bf16* Wt, const float* bias,
                  float* C, int N, int shift, int accum) {
    gemm_bf16_wmma<<<dim3(TP / 128, N / 128), 256, 0, stream>>>(
        A, Wt, bias, C, ZPAGE, TP, K, N, shift, accum);
  };
  auto tcvt = [&](const float* src, __bf16* dst, int K, int N) {
    transpose_cvt_kernel<<<dim3(K / 32, N / 32), 256, 0, stream>>>(src, dst, K, N);
  };

  for (int k = 11; k >= 0; --k) {
    const int nh   = (k >= 8) ? 2 : (k >= 4) ? 3 : 4;
    const int nrem = 2 * nh;
    const int base = fb(k);

    // convert this flow's big weights: cond(640x512), in taps(256x512 x3), rs(256xNrs)
    for (int i = 0; i < 8; ++i) {
      __bf16* lb = WSLAB + (size_t)i * 851968;
      tcvt(F(base + 8 + i), lb, CMEL, 512);                                  // cond_w
      for (int j = 0; j < 3; ++j)
        tcvt(F(base + 26 + i) + (size_t)j * NC * 512, lb + 327680 + (size_t)j * 131072, NC, 512);
      tcvt(F(base + 43 + i), lb + 720896, NC, (i < 7) ? 512 : 256);          // rs_w
    }

    // x = start(a0)
    start_kernel<<<blocks1d((size_t)TP * NC), 256, 0, stream>>>(
        AUDIO, F(base + 52), F(base + 51), nh, X_F, X_BF);

    for (int i = 0; i < 8; ++i) {
      const int d = 1 << i;
      __bf16* lb = WSLAB + (size_t)i * 851968;
      // z = s @ cond_w + cond_b
      gemm(S_BF, CMEL, lb, F(base + i), Z, 512, 0, 0);
      // z += dilated conv taps (SAME): out[t] += x[t+(h-1)d] @ in_w[h]; in_b on tap 0
      gemm(X_BF, NC, lb + 327680 + 0 * 131072, F(base + 18 + i), Z, 512, -d, 1);
      gemm(X_BF, NC, lb + 327680 + 1 * 131072, nullptr,          Z, 512,  0, 1);
      gemm(X_BF, NC, lb + 327680 + 2 * 131072, nullptr,          Z, 512, +d, 1);
      // acts = tanh(z_lo) * sigmoid(z_hi)
      gate_kernel<<<blocks1d((size_t)TP * NC), 256, 0, stream>>>(Z, ACT_BF);
      // rs = acts @ rs_w + rs_b
      const int Nrs = (i < 7) ? 512 : 256;
      gemm(ACT_BF, NC, lb + 720896, F(base + 35 + i), RS, Nrs, 0, 0);
      // x += rs_lo ; out (+)= rs_hi (or rs on last layer)
      resskip_kernel<<<blocks1d((size_t)TP * NC), 256, 0, stream>>>(
          RS, X_F, X_BF, OUTF, (i == 0) ? 1 : 0, (i == 7) ? 1 : 0);
    }

    // o = out @ end_w + end_b ; affine coupling on a1
    end_kernel<<<blocks1d((size_t)TP * 2 * nh), 256, 0, stream>>>(
        OUTF, F(base + 17), F(base + 16), 2 * nh, OEND);
    coupling_kernel<<<blocks1d((size_t)TP * nh), 256, 0, stream>>>(AUDIO, OEND, nh);

    // audio = audio @ inv(inv_w), then optional z_early prepend
    invert_kernel<<<1, 32, 0, stream>>>(F(base + 34), INVM, nrem);
    const int pre = (k % 4 == 0 && k > 0) ? 1 : 0;
    permute_kernel<<<blocks1d(TP), 256, 0, stream>>>(
        AUDIO, INVM, nrem, (k == 8) ? ze8 : ze4, pre);
  }

  final_copy_kernel<<<blocks1d((size_t)TP * 8), 256, 0, stream>>>(AUDIO, (float*)d_out);
}
